// AttLIF_57148834841061
// MI455X (gfx1250) — matrix operand order repaired and checked by reading, not compiled, not run on hardware
//
#include <hip/hip_runtime.h>
#include <hip/hip_bf16.h>

typedef __attribute__((ext_vector_type(16))) __bf16 v16bf;
typedef __attribute__((ext_vector_type(8)))  float  v8f;

#define B_DIM 128
#define T_DIM 64
#define I_DIM 2048
#define H_DIM 2048
#define M_DIM (B_DIM * T_DIM)   // 8192 tokens
#define LDSK  40                // padded LDS row stride (halves): 80B, keeps 16B alignment

#define ALPHA 0.3f
#define VTH   0.3f

// ws layout (big-ws path): [gate: 8192 f32][Ahi: M*K u16][Alo][Whi: H*K u16][Wlo]
#define GATE_BYTES   32768
#define APLANE_ELEMS ((size_t)M_DIM * I_DIM)
#define WPLANE_ELEMS ((size_t)H_DIM * I_DIM)
#define WS_NEEDED    (GATE_BYTES + (APLANE_ELEMS * 2 + WPLANE_ELEMS * 2) * sizeof(unsigned short))

// ---------- fp32 -> bf16 (RNE) and hi/lo split for bf16x3 emulated-fp32 GEMM ----------
__device__ __forceinline__ unsigned short bf16_rn(float f) {
  unsigned int x = __float_as_uint(f);
  x += 0x7FFFu + ((x >> 16) & 1u);
  return (unsigned short)(x >> 16);
}
__device__ __forceinline__ void split2(float f, unsigned short& hi, unsigned short& lo) {
  hi = bf16_rn(f);
  float fh = __uint_as_float((unsigned int)hi << 16);
  lo = bf16_rn(f - fh);
}

// ---------- CDNA5 async copy: 16B global -> LDS, tracked by ASYNCcnt ----------
__device__ __forceinline__ void async_cp16(unsigned int lds_off, const void* gptr) {
  asm volatile("global_load_async_to_lds_b128 %0, %1, off"
               :: "v"(lds_off), "v"((unsigned long long)(uintptr_t)gptr)
               : "memory");
}
#define S_WAIT_ASYNC6() asm volatile("s_wait_asynccnt 0x6" ::: "memory")
#define S_WAIT_ASYNC0() asm volatile("s_wait_asynccnt 0x0" ::: "memory")

// ---------- Kernel 0: one-shot fp32 -> (bf16 hi, bf16 lo) plane split ----------
__global__ __launch_bounds__(256)
void preconv_kernel(const float* __restrict__ src,
                    unsigned short* __restrict__ hi,
                    unsigned short* __restrict__ lo, int n4) {
  int i = blockIdx.x * 256 + threadIdx.x;
  if (i >= n4) return;
  float4 v = ((const float4*)src)[i];
  ushort4 h, l;
  split2(v.x, h.x, l.x);
  split2(v.y, h.y, l.y);
  split2(v.z, h.z, l.z);
  split2(v.w, h.w, l.w);
  ((ushort4*)hi)[i] = h;
  ((ushort4*)lo)[i] = l;
}

// ---------- Kernel 1a: GEMM from pre-split bf16 planes, async-LDS double buffered ----------
// Workgroup tile: 128 (M) x 64 (N). 8 waves (4 along M x 2 along N), each wave 32x32.
__global__ __launch_bounds__(256)
void gemm_async_kernel(const unsigned short* __restrict__ Ahi,
                       const unsigned short* __restrict__ Alo,
                       const unsigned short* __restrict__ Whi,
                       const unsigned short* __restrict__ Wlo,
                       const float* __restrict__ bias,
                       float* __restrict__ X) {
  __shared__ unsigned short sA[2][2][128 * LDSK]; // [buf][hi/lo]
  __shared__ unsigned short sB[2][2][64 * LDSK];

  const int tid    = threadIdx.x;
  const int blockM = blockIdx.x * 128;
  const int blockN = blockIdx.y * 64;
  const int wid    = tid >> 5;
  const int lane   = tid & 31;
  const int waveM  = (wid & 3) * 32;
  const int waveN  = (wid >> 2) * 32;
  const int half   = lane >> 4;
  const int lr     = lane & 15;

  v8f acc[2][2] = {};

  // Issue one double-buffer batch: 6 async B128 per thread (A:4, B:2)
  auto issue = [&](int buf, int kb) {
    #pragma unroll
    for (int pl = 0; pl < 2; ++pl) {
      const unsigned short* ga = pl ? Alo : Ahi;
      unsigned short* la = &sA[buf][pl][0];
      #pragma unroll
      for (int it = 0; it < 2; ++it) {
        int c   = tid + it * 256;   // 512 chunks = 128 rows x 4
        int row = c >> 2;
        int col = (c & 3) * 8;
        async_cp16((unsigned int)(uintptr_t)(la + row * LDSK + col),
                   ga + (size_t)(blockM + row) * I_DIM + kb + col);
      }
      const unsigned short* gb = pl ? Wlo : Whi;
      unsigned short* lb = &sB[buf][pl][0];
      int c   = tid;                // 256 chunks = 64 rows x 4
      int row = c >> 2;
      int col = (c & 3) * 8;
      async_cp16((unsigned int)(uintptr_t)(lb + row * LDSK + col),
                 gb + (size_t)(blockN + row) * I_DIM + kb + col);
    }
  };

  issue(0, 0);
  for (int kb = 0, i = 0; kb < I_DIM; kb += 32, ++i) {
    const int cur = i & 1;
    if (kb + 32 < I_DIM) { issue(cur ^ 1, kb + 32); S_WAIT_ASYNC6(); }
    else                 { S_WAIT_ASYNC0(); }
    __syncthreads();   // all waves' async writes for slab `cur` have landed

    // A-operand (16x32): lane<16 row=lr K{0..7,16..23}; lane>=16 K{8..15,24..31}
    v16bf ahi[2], alo[2], bhi[2], blo[2];
    #pragma unroll
    for (int mi = 0; mi < 2; ++mi) {
      const unsigned short* ph = &sA[cur][0][(waveM + mi * 16 + lr) * LDSK];
      const unsigned short* pl = &sA[cur][1][(waveM + mi * 16 + lr) * LDSK];
      union { v16bf v; uint4 q[2]; } u;
      u.q[0] = *(const uint4*)(ph + half * 8);
      u.q[1] = *(const uint4*)(ph + 16 + half * 8);
      ahi[mi] = u.v;
      u.q[0] = *(const uint4*)(pl + half * 8);
      u.q[1] = *(const uint4*)(pl + 16 + half * 8);
      alo[mi] = u.v;
    }
    // B-operand (32x16): lane<16 col=lr K{0..7,16..23}; lane>=16 K{8..15,24..31}
    #pragma unroll
    for (int ni = 0; ni < 2; ++ni) {
      const unsigned short* ph = &sB[cur][0][(waveN + ni * 16 + lr) * LDSK];
      const unsigned short* pl = &sB[cur][1][(waveN + ni * 16 + lr) * LDSK];
      union { v16bf v; uint4 q[2]; } u;
      u.q[0] = *(const uint4*)(ph + half * 8);
      u.q[1] = *(const uint4*)(ph + 16 + half * 8);
      bhi[ni] = u.v;
      u.q[0] = *(const uint4*)(pl + half * 8);
      u.q[1] = *(const uint4*)(pl + 16 + half * 8);
      blo[ni] = u.v;
    }

    #pragma unroll
    for (int mi = 0; mi < 2; ++mi)
      #pragma unroll
      for (int ni = 0; ni < 2; ++ni) {
        acc[mi][ni] = __builtin_amdgcn_wmma_f32_16x16x32_bf16(
            false, ahi[mi], false, bhi[ni], (short)0, acc[mi][ni], false, false);
        acc[mi][ni] = __builtin_amdgcn_wmma_f32_16x16x32_bf16(
            false, ahi[mi], false, blo[ni], (short)0, acc[mi][ni], false, false);
        acc[mi][ni] = __builtin_amdgcn_wmma_f32_16x16x32_bf16(
            false, alo[mi], false, bhi[ni], (short)0, acc[mi][ni], false, false);
      }
    __syncthreads();   // done reading slab `cur` before it is overwritten
  }

  #pragma unroll
  for (int mi = 0; mi < 2; ++mi)
    #pragma unroll
    for (int ni = 0; ni < 2; ++ni) {
      int col  = blockN + waveN + ni * 16 + lr;
      float bv = bias[col];
      #pragma unroll
      for (int r = 0; r < 8; ++r) {
        int row = blockM + waveM + mi * 16 + half * 8 + r;
        X[(size_t)row * H_DIM + col] = acc[mi][ni][r] + bv;
      }
    }
}

// ---------- Kernel 1b (fallback, small ws): fused split + GEMM ----------
__global__ __launch_bounds__(256)
void gemm_bf16x3_kernel(const float* __restrict__ A,
                        const float* __restrict__ Wm,
                        const float* __restrict__ bias,
                        float* __restrict__ X) {
  __shared__ unsigned short sAhi[128 * LDSK];
  __shared__ unsigned short sAlo[128 * LDSK];
  __shared__ unsigned short sBhi[64 * LDSK];
  __shared__ unsigned short sBlo[64 * LDSK];

  const int tid    = threadIdx.x;
  const int blockM = blockIdx.x * 128;
  const int blockN = blockIdx.y * 64;
  const int wid    = tid >> 5;
  const int lane   = tid & 31;
  const int waveM  = (wid & 3) * 32;
  const int waveN  = (wid >> 2) * 32;
  const int half   = lane >> 4;
  const int lr     = lane & 15;

  v8f acc[2][2] = {};

  for (int kb = 0; kb < I_DIM; kb += 32) {
    __syncthreads();
    #pragma unroll
    for (int it = 0; it < 4; ++it) {
      int i   = tid + it * 256;
      int row = i >> 3;
      int c4  = (i & 7) << 2;
      const float* src = A + (size_t)(blockM + row) * I_DIM + kb + c4;
      float4 v = *(const float4*)src;
      if (kb + 32 < I_DIM) __builtin_prefetch(src + 32, 0, 0);
      int o = row * LDSK + c4;
      unsigned short h, l;
      split2(v.x, h, l); sAhi[o + 0] = h; sAlo[o + 0] = l;
      split2(v.y, h, l); sAhi[o + 1] = h; sAlo[o + 1] = l;
      split2(v.z, h, l); sAhi[o + 2] = h; sAlo[o + 2] = l;
      split2(v.w, h, l); sAhi[o + 3] = h; sAlo[o + 3] = l;
    }
    #pragma unroll
    for (int it = 0; it < 2; ++it) {
      int i   = tid + it * 256;
      int row = i >> 3;
      int c4  = (i & 7) << 2;
      const float* src = Wm + (size_t)(blockN + row) * I_DIM + kb + c4;
      float4 v = *(const float4*)src;
      if (kb + 32 < I_DIM) __builtin_prefetch(src + 32, 0, 0);
      int o = row * LDSK + c4;
      unsigned short h, l;
      split2(v.x, h, l); sBhi[o + 0] = h; sBlo[o + 0] = l;
      split2(v.y, h, l); sBhi[o + 1] = h; sBlo[o + 1] = l;
      split2(v.z, h, l); sBhi[o + 2] = h; sBlo[o + 2] = l;
      split2(v.w, h, l); sBhi[o + 3] = h; sBlo[o + 3] = l;
    }
    __syncthreads();

    v16bf ahi[2], alo[2], bhi[2], blo[2];
    #pragma unroll
    for (int mi = 0; mi < 2; ++mi) {
      const unsigned short* ph = &sAhi[(waveM + mi * 16 + lr) * LDSK];
      const unsigned short* pl = &sAlo[(waveM + mi * 16 + lr) * LDSK];
      union { v16bf v; uint4 q[2]; } u;
      u.q[0] = *(const uint4*)(ph + half * 8);
      u.q[1] = *(const uint4*)(ph + 16 + half * 8);
      ahi[mi] = u.v;
      u.q[0] = *(const uint4*)(pl + half * 8);
      u.q[1] = *(const uint4*)(pl + 16 + half * 8);
      alo[mi] = u.v;
    }
    #pragma unroll
    for (int ni = 0; ni < 2; ++ni) {
      const unsigned short* ph = &sBhi[(waveN + ni * 16 + lr) * LDSK];
      const unsigned short* pl = &sBlo[(waveN + ni * 16 + lr) * LDSK];
      union { v16bf v; uint4 q[2]; } u;
      u.q[0] = *(const uint4*)(ph + half * 8);
      u.q[1] = *(const uint4*)(ph + 16 + half * 8);
      bhi[ni] = u.v;
      u.q[0] = *(const uint4*)(pl + half * 8);
      u.q[1] = *(const uint4*)(pl + 16 + half * 8);
      blo[ni] = u.v;
    }

    #pragma unroll
    for (int mi = 0; mi < 2; ++mi)
      #pragma unroll
      for (int ni = 0; ni < 2; ++ni) {
        acc[mi][ni] = __builtin_amdgcn_wmma_f32_16x16x32_bf16(
            false, ahi[mi], false, bhi[ni], (short)0, acc[mi][ni], false, false);
        acc[mi][ni] = __builtin_amdgcn_wmma_f32_16x16x32_bf16(
            false, ahi[mi], false, blo[ni], (short)0, acc[mi][ni], false, false);
        acc[mi][ni] = __builtin_amdgcn_wmma_f32_16x16x32_bf16(
            false, alo[mi], false, bhi[ni], (short)0, acc[mi][ni], false, false);
      }
  }

  #pragma unroll
  for (int mi = 0; mi < 2; ++mi)
    #pragma unroll
    for (int ni = 0; ni < 2; ++ni) {
      int col  = blockN + waveN + ni * 16 + lr;
      float bv = bias[col];
      #pragma unroll
      for (int r = 0; r < 8; ++r) {
        int row = blockM + waveM + mi * 16 + half * 8 + r;
        X[(size_t)row * H_DIM + col] = acc[mi][ni][r] + bv;
      }
    }
}

// ---------- Kernel 2: temporal-attention gate a[b,t] ----------
__global__ __launch_bounds__(256)
void ta_kernel(const float* __restrict__ x,
               const float* __restrict__ w1, const float* __restrict__ b1,
               const float* __restrict__ w2, const float* __restrict__ b2,
               float* __restrict__ a_out) {
  __shared__ float s_sh[T_DIM];
  __shared__ float r_sh[4];
  __shared__ float wpart[8];
  const int b   = blockIdx.x;
  const int tid = threadIdx.x;

  for (int t = 0; t < T_DIM; ++t) {
    const float4* row = (const float4*)(x + (size_t)(b * T_DIM + t) * H_DIM);
    float4 v0 = row[tid * 2 + 0];
    float4 v1 = row[tid * 2 + 1];
    float v = v0.x + v0.y + v0.z + v0.w + v1.x + v1.y + v1.z + v1.w;
    #pragma unroll
    for (int off = 16; off > 0; off >>= 1) v += __shfl_xor(v, off, 32);
    if ((tid & 31) == 0) wpart[tid >> 5] = v;
    __syncthreads();
    if (tid == 0) {
      float tot = 0.f;
      #pragma unroll
      for (int w = 0; w < 8; ++w) tot += wpart[w];
      s_sh[t] = tot * (1.0f / (float)H_DIM);
    }
    __syncthreads();
  }
  if (tid < 4) {
    float acc = b1[tid];
    #pragma unroll 8
    for (int t = 0; t < T_DIM; ++t) acc += s_sh[t] * w1[tid * T_DIM + t];
    r_sh[tid] = fmaxf(acc, 0.f);
  }
  __syncthreads();
  if (tid < T_DIM) {
    float p = b2[tid];
    #pragma unroll
    for (int j = 0; j < 4; ++j) p += r_sh[j] * w2[tid * 4 + j];
    a_out[b * T_DIM + tid] = 1.f / (1.f + __expf(-p));
  }
}

// ---------- Kernel 3: gate * x, then LIF scan over T, in place on d_out ----------
__global__ __launch_bounds__(256)
void lif_kernel(float* __restrict__ xio, const float* __restrict__ a_g) {
  __shared__ float a_sh[T_DIM];
  const int b   = blockIdx.y;
  const int tid = threadIdx.x;
  if (tid < T_DIM) a_sh[tid] = a_g[b * T_DIM + tid];
  __syncthreads();

  const int h4 = blockIdx.x * 256 + tid;
  float4* base = (float4*)xio;
  size_t idx = (size_t)b * T_DIM * (H_DIM / 4) + h4;
  float4 u = {0.f, 0.f, 0.f, 0.f};
  for (int t = 0; t < T_DIM; ++t) {
    float4 xt = base[idx];
    float g = a_sh[t];
    float4 sp;
    u.x = ALPHA * u.x + xt.x * g; sp.x = (u.x >= VTH) ? 1.f : 0.f; u.x *= (1.f - sp.x);
    u.y = ALPHA * u.y + xt.y * g; sp.y = (u.y >= VTH) ? 1.f : 0.f; u.y *= (1.f - sp.y);
    u.z = ALPHA * u.z + xt.z * g; sp.z = (u.z >= VTH) ? 1.f : 0.f; u.z *= (1.f - sp.z);
    u.w = ALPHA * u.w + xt.w * g; sp.w = (u.w >= VTH) ? 1.f : 0.f; u.w *= (1.f - sp.w);
    base[idx] = sp;
    idx += H_DIM / 4;
  }
}

extern "C" void kernel_launch(void* const* d_in, const int* in_sizes, int n_in,
                              void* d_out, int out_size, void* d_ws, size_t ws_size,
                              hipStream_t stream) {
  const float* data = (const float*)d_in[0]; // [128,64,2048]
  const float* W    = (const float*)d_in[1]; // [2048,2048]
  const float* bias = (const float*)d_in[2]; // [2048]
  const float* w1   = (const float*)d_in[3]; // [4,64]
  const float* b1   = (const float*)d_in[4]; // [4]
  const float* w2   = (const float*)d_in[5]; // [64,4]
  const float* b2   = (const float*)d_in[6]; // [64]
  float* out    = (float*)d_out;             // x scratch then spikes [128,64,2048]
  float* a_gate = (float*)d_ws;              // [128,64]

  dim3 g1(M_DIM / 128, H_DIM / 64);          // 64 x 32 workgroups

  if (ws_size >= WS_NEEDED) {
    unsigned short* Ahi = (unsigned short*)((char*)d_ws + GATE_BYTES);
    unsigned short* Alo = Ahi + APLANE_ELEMS;
    unsigned short* Whi = Alo + APLANE_ELEMS;
    unsigned short* Wlo = Whi + WPLANE_ELEMS;

    int nA4 = (int)(APLANE_ELEMS / 4);       // 4,194,304
    int nW4 = (int)(WPLANE_ELEMS / 4);       // 1,048,576
    preconv_kernel<<<(nA4 + 255) / 256, 256, 0, stream>>>(data, Ahi, Alo, nA4);
    preconv_kernel<<<(nW4 + 255) / 256, 256, 0, stream>>>(W, Whi, Wlo, nW4);
    gemm_async_kernel<<<g1, 256, 0, stream>>>(Ahi, Alo, Whi, Wlo, bias, out);
  } else {
    gemm_bf16x3_kernel<<<g1, 256, 0, stream>>>(data, W, bias, out);
  }

  ta_kernel<<<B_DIM, 256, 0, stream>>>(out, w1, b1, w2, b2, a_gate);

  dim3 g3(H_DIM / (4 * 256), B_DIM);         // 2 x 128
  lif_kernel<<<g3, 256, 0, stream>>>(out, a_gate);
}
